// ConvocationV3_2284922601699
// MI455X (gfx1250) — compile-verified, matrix operand order checked by reading, not run on hardware
//
#include <hip/hip_runtime.h>
#include <hip/hip_bf16.h>
#include <math.h>

// ---------------- problem constants ----------------
#define BATCH 32
#define CH    384
#define DQ    96
#define HH    54
#define WW    54
#define HW    (HH*WW)      // 2916
#define NKT   9            // 3x3 kernel taps

typedef __bf16 bf16_t;
typedef bf16_t v16bf __attribute__((ext_vector_type(16)));
typedef bf16_t v8bf  __attribute__((ext_vector_type(8)));
typedef bf16_t v4bf  __attribute__((ext_vector_type(4)));
typedef float  v8f   __attribute__((ext_vector_type(8)));

// ---------------- CDNA5 async copy helpers ----------------
// GLOBAL_LOAD_ASYNC_TO_LDS_B128: per-lane 16B global->LDS copy, tracked by
// ASYNCcnt (CDNA5 ISA). LDS address = low 32 bits of
// the generic pointer (LDS aperture keeps the byte offset in addr[31:0]).
__device__ __forceinline__ void async_copy_b128(uint32_t lds_off, uint64_t gaddr) {
    asm volatile("global_load_async_to_lds_b128 %0, %1, off"
                 :: "v"(lds_off), "v"(gaddr) : "memory");
}
__device__ __forceinline__ void async_wait0() {
    asm volatile("s_wait_asynccnt 0x0" ::: "memory");
}

// ======================================================================
// Kernel 1: 18x18 average pool of x -> xp (B,C,3,3), fused f32->bf16
// copy of x (x is streamed here anyway; the value GEMM then reads bf16).
// ======================================================================
__global__ __launch_bounds__(128)
void pool18_kernel(const float* __restrict__ x, float* __restrict__ xp,
                   bf16_t* __restrict__ xbf) {
    const int bc  = blockIdx.x;          // 0 .. B*C-1
    const int tid = threadIdx.x;         // 128 threads
    const float* src = x + (size_t)bc * HW;
    bf16_t* dst = xbf + (size_t)bc * HW;
    float acc[NKT];
    #pragma unroll
    for (int i = 0; i < NKT; ++i) acc[i] = 0.f;
    for (int p = tid; p < HW; p += 128) {
        float v = src[p];
        dst[p] = (bf16_t)v;
        int h = p / WW, w = p - h * WW;
        acc[(h / 18) * 3 + (w / 18)] += v;
    }
    __shared__ float red[NKT][128];
    #pragma unroll
    for (int i = 0; i < NKT; ++i) red[i][tid] = acc[i];
    __syncthreads();
    for (int s = 64; s > 0; s >>= 1) {
        if (tid < s) {
            #pragma unroll
            for (int i = 0; i < NKT; ++i) red[i][tid] += red[i][tid + s];
        }
        __syncthreads();
    }
    if (tid == 0) {
        const float inv = 1.0f / 324.0f;
        #pragma unroll
        for (int i = 0; i < NKT; ++i) xp[(size_t)bc * NKT + i] = red[i][0] * inv;
    }
}

// ======================================================================
// Kernel 1b: one-time f32->bf16 conversion of the two big GEMM weights
// ======================================================================
__global__ __launch_bounds__(256)
void cvtw_kernel(const float* __restrict__ w_v, const float* __restrict__ w_proj,
                 bf16_t* __restrict__ wv_bf, bf16_t* __restrict__ wp_bf) {
    int i = blockIdx.x * 256 + threadIdx.x;
    if (i < CH * CH) {
        wv_bf[i] = (bf16_t)w_v[i];
        wp_bf[i] = (bf16_t)w_proj[i];
    }
}

// ======================================================================
// Kernel 2: per-batch kernel-generation MLP (tiny: 9 pixels per batch)
// ======================================================================
__global__ __launch_bounds__(CH)
void kgen_kernel(const float* __restrict__ xp,
                 const float* __restrict__ w_qk, const float* __restrict__ b_qk,
                 const float* __restrict__ w_kg1, const float* __restrict__ b_kg1,
                 const float* __restrict__ w_kg2, const float* __restrict__ b_kg2,
                 const float* __restrict__ beta, float* __restrict__ kern) {
    const int b = blockIdx.x;
    const int o = threadIdx.x;           // 384 threads
    __shared__ float sxp[CH * NKT];
    __shared__ float sqk[CH * NKT];
    __shared__ float shh[DQ * NKT];
    for (int i = o; i < CH * NKT; i += CH) sxp[i] = xp[(size_t)b * CH * NKT + i];
    __syncthreads();
    {   // qk = W_qk @ xp + b_qk (pool-before-1x1: mathematically identical)
        float acc[NKT]; float bb = b_qk[o];
        #pragma unroll
        for (int j = 0; j < NKT; ++j) acc[j] = bb;
        const float* wr = w_qk + (size_t)o * CH;
        for (int c = 0; c < CH; ++c) {
            float wv = wr[c];
            const float* xr = &sxp[c * NKT];
            #pragma unroll
            for (int j = 0; j < NKT; ++j) acc[j] += wv * xr[j];
        }
        #pragma unroll
        for (int j = 0; j < NKT; ++j) sqk[o * NKT + j] = acc[j];
    }
    __syncthreads();
    if (o < DQ) {   // h = gelu_exact(W_kg1 @ qk + b_kg1)
        float acc[NKT]; float bb = b_kg1[o];
        #pragma unroll
        for (int j = 0; j < NKT; ++j) acc[j] = bb;
        const float* wr = w_kg1 + (size_t)o * CH;
        for (int c = 0; c < CH; ++c) {
            float wv = wr[c];
            const float* xr = &sqk[c * NKT];
            #pragma unroll
            for (int j = 0; j < NKT; ++j) acc[j] += wv * xr[j];
        }
        #pragma unroll
        for (int j = 0; j < NKT; ++j) {
            float p = acc[j];
            shh[o * NKT + j] = 0.5f * p * (1.0f + erff(p * 0.70710678118654752f));
        }
    }
    __syncthreads();
    {   // kern = W_kg2 @ h + b_kg2 ; k -= sigmoid(beta) * mean3x3(k)
        float acc[NKT]; float bb = b_kg2[o];
        #pragma unroll
        for (int j = 0; j < NKT; ++j) acc[j] = bb;
        const float* wr = w_kg2 + (size_t)o * DQ;
        for (int c = 0; c < DQ; ++c) {
            float wv = wr[c];
            const float* xr = &shh[c * NKT];
            #pragma unroll
            for (int j = 0; j < NKT; ++j) acc[j] += wv * xr[j];
        }
        float mean = 0.f;
        #pragma unroll
        for (int j = 0; j < NKT; ++j) mean += acc[j];
        mean *= (1.0f / 9.0f);
        float f = 1.0f / (1.0f + expf(-beta[o]));
        float* dst = kern + ((size_t)b * CH + o) * NKT;
        #pragma unroll
        for (int j = 0; j < NKT; ++j) dst[j] = acc[j] - f * mean;
    }
}

// ======================================================================
// bf16 WMMA GEMM tiles: 128x128x32, 256 threads = 8 waves, wave = 32x64,
// 2-stage LDS double buffering (stage k+1 overlaps compute of k).
// ======================================================================
#define BM 128
#define BN 128
#define BK 32
#define LDK 40   // padded LDS row stride (80B: 16B-aligned, conflict-light)
#define NTK (CH / BK)   // 12 K-steps

// 16x32 bf16 fragment (A-layout, ISA 7.12.2) from an LDS [row][LDK] region:
// lane 0-15 = row M with K 0..7 / 16..23; lanes 16-31 same rows, K 8..15 /
// 24..31. Two aligned 16B LDS reads + shuffle -> paired ds_load_b128.
__device__ __forceinline__ v16bf load_frag(const bf16_t* rowp, int kbase) {
    v8bf lo = *(const v8bf*)(rowp + kbase);
    v8bf hi = *(const v8bf*)(rowp + kbase + 16);
    return __builtin_shufflevector(lo, hi, 0, 1, 2, 3, 4, 5, 6, 7,
                                           8, 9, 10, 11, 12, 13, 14, 15);
}

// Async-stage a 128x32 bf16 A tile (row-major, contiguous 64B rows) into
// LDS via GLOBAL_LOAD_ASYNC_TO_LDS_B128 (2 chunks of 16B per thread).
__device__ __forceinline__ void stage_A_async(const bf16_t* __restrict__ Abf,
                                              int m0, int kt,
                                              bf16_t (*As)[LDK], int tid) {
    const uint32_t base = (uint32_t)(uintptr_t)(&As[0][0]);
    #pragma unroll
    for (int it = 0; it < 2; ++it) {
        int s   = tid + it * 256;     // 0..511 chunks of 16B
        int row = s >> 2;             // 4 chunks per 32-element row
        int c8  = (s & 3) << 3;       // element offset within row
        uint64_t g = (uint64_t)(uintptr_t)(Abf + (size_t)(m0 + row) * CH + kt + c8);
        uint32_t l = base + (uint32_t)(row * LDK + c8) * 2u;
        async_copy_b128(l, g);
    }
}

// Stage a 32x128 bf16 B tile from [k][n] global into [n][k] LDS (8B chunks).
__device__ __forceinline__ void stage_B_value(const bf16_t* __restrict__ Bp,
                                              int n0, int kt,
                                              bf16_t (*Bs)[LDK], int tid) {
    for (int s = tid; s < (BK * BN) / 4; s += 256) {
        int krow = s >> 5;            // 32 4-elem chunks per 128-wide row
        int n4   = (s & 31) << 2;
        int gn   = n0 + n4;
        v4bf v = {};
        if (gn + 3 < HW)              // HW%4==0, chunk all-in or all-out
            v = *(const v4bf*)(Bp + (size_t)(kt + krow) * HW + gn);
        Bs[n4 + 0][krow] = v[0];
        Bs[n4 + 1][krow] = v[1];
        Bs[n4 + 2][krow] = v[2];
        Bs[n4 + 3][krow] = v[3];
    }
}

// Stage a 32x128 B tile computing the per-sample depthwise 3x3 on the fly.
__device__ __forceinline__ void stage_B_dw(const bf16_t* __restrict__ value,
                                           const float* __restrict__ kern,
                                           int bat, int n0, int kt,
                                           bf16_t (*Bs)[LDK], int tid) {
    for (int s = tid; s < BK * BN; s += 256) {
        int krow = s >> 7;            // local k (channel)
        int n    = s & 127;
        int gn   = n0 + n;            // flattened pixel
        float r = 0.f;
        if (gn < HW) {
            int c = kt + krow;
            const bf16_t* vsrc = value + ((size_t)bat * CH + c) * HW;
            const float*  kp   = kern + ((size_t)bat * CH + c) * NKT;
            int h = gn / WW, w = gn - h * WW;
            #pragma unroll
            for (int dy = -1; dy <= 1; ++dy) {
                int hh = h + dy;
                if ((unsigned)hh < (unsigned)HH) {
                    const bf16_t* rowv = vsrc + hh * WW;
                    #pragma unroll
                    for (int dx = -1; dx <= 1; ++dx) {
                        int ww2 = w + dx;
                        if ((unsigned)ww2 < (unsigned)WW)
                            r += kp[(dy + 1) * 3 + (dx + 1)] * (float)rowv[ww2];
                    }
                }
            }
        }
        Bs[n][krow] = (bf16_t)r;
    }
}

// 8 WMMAs of one K-step for this wave (2x4 16x16 tiles of its 32x64 patch).
__device__ __forceinline__ void wmma_step(bf16_t (*As)[LDK], bf16_t (*Bs)[LDK],
                                          v8f acc[2][4], int wrow, int wcol,
                                          int l15, int kbase) {
    v16bf afrag[2], bfrag[4];
    #pragma unroll
    for (int i = 0; i < 2; ++i)
        afrag[i] = load_frag(&As[wrow * 32 + i * 16 + l15][0], kbase);
    #pragma unroll
    for (int j = 0; j < 4; ++j)
        bfrag[j] = load_frag(&Bs[wcol * 64 + j * 16 + l15][0], kbase);
    #pragma unroll
    for (int i = 0; i < 2; ++i)
        #pragma unroll
        for (int j = 0; j < 4; ++j)
            acc[i][j] = __builtin_amdgcn_wmma_f32_16x16x32_bf16(
                false, afrag[i], false, bfrag[j],
                (short)0, acc[i][j], false, false);
}

// ----------------------------------------------------------------------
// value = w_v(bf16) @ xbf(bf16) + b_v  -> value (bf16 scratch)
// ----------------------------------------------------------------------
__global__ __launch_bounds__(256)
void gemm_value_kernel(const bf16_t* __restrict__ Abf,  // w_v bf16 (CHxCH)
                       const bf16_t* __restrict__ Bbf,  // xbf (B,CH,HW)
                       const float* __restrict__ bias,  // b_v
                       bf16_t* __restrict__ C) {        // value (B,CH,HW)
    const int n0  = blockIdx.x * BN;
    const int m0  = blockIdx.y * BM;
    const int bat = blockIdx.z;
    const bf16_t* Bp = Bbf + (size_t)bat * CH * HW;
    bf16_t* Cp = C + (size_t)bat * CH * HW;

    __shared__ __align__(16) bf16_t As[2][BM][LDK];
    __shared__ __align__(16) bf16_t Bs[2][BN][LDK];

    const int tid  = threadIdx.x;
    const int lane = tid & 31;
    const int wave = tid >> 5;
    const int wrow = wave >> 1;
    const int wcol = wave & 1;
    const int l15  = lane & 15;
    const int kbase = (lane >> 4) * 8;

    v8f acc[2][4];
    #pragma unroll
    for (int i = 0; i < 2; ++i)
        #pragma unroll
        for (int j = 0; j < 4; ++j) acc[i][j] = {};

    // prologue: stage tile 0
    stage_A_async(Abf, m0, 0, As[0], tid);
    stage_B_value(Bp, n0, 0, Bs[0], tid);
    async_wait0();
    __syncthreads();

    for (int it = 0; it < NTK; ++it) {
        const int cur = it & 1, nxt = cur ^ 1;
        if (it + 1 < NTK) {   // stage next tile first: overlaps with WMMAs
            __builtin_prefetch(Bp + (size_t)(it + 1) * BK * HW + n0, 0, 1);
            stage_A_async(Abf, m0, (it + 1) * BK, As[nxt], tid);
            stage_B_value(Bp, n0, (it + 1) * BK, Bs[nxt], tid);
        }
        wmma_step(As[cur], Bs[cur], acc, wrow, wcol, l15, kbase);
        async_wait0();        // next-tile async copies landed
        __syncthreads();      // everyone done reading cur / writing nxt
    }

    // C layout (ISA): vgpr r -> M = r + 8*(lane>=16), N = lane&15
    #pragma unroll
    for (int i = 0; i < 2; ++i) {
        int gmBase = m0 + wrow * 32 + i * 16 + (lane >> 4) * 8;
        #pragma unroll
        for (int j = 0; j < 4; ++j) {
            int gn = n0 + wcol * 64 + j * 16 + l15;
            if (gn < HW) {
                #pragma unroll
                for (int r = 0; r < 8; ++r) {
                    int gm = gmBase + r;
                    Cp[(size_t)gm * HW + gn] = (bf16_t)(acc[i][j][r] + bias[gm]);
                }
            }
        }
    }
}

// ----------------------------------------------------------------------
// out = w_proj @ depthwise3x3(value, kern) + b_proj, depthwise fused into
// the (double-buffered) B-tile stage; its 9-tap global reads hide under
// the WMMA block of the previous tile.
// ----------------------------------------------------------------------
__global__ __launch_bounds__(256)
void gemm_dwproj_kernel(const bf16_t* __restrict__ Abf,   // w_proj bf16
                        const bf16_t* __restrict__ value, // (B,CH,HW) bf16
                        const float* __restrict__ kern,   // (B,CH,9)
                        const float* __restrict__ bias,   // b_proj
                        float* __restrict__ C) {          // (B,CH,HW) f32
    const int n0  = blockIdx.x * BN;
    const int m0  = blockIdx.y * BM;
    const int bat = blockIdx.z;
    float* Cp = C + (size_t)bat * CH * HW;

    __shared__ __align__(16) bf16_t As[2][BM][LDK];
    __shared__ __align__(16) bf16_t Bs[2][BN][LDK];

    const int tid  = threadIdx.x;
    const int lane = tid & 31;
    const int wave = tid >> 5;
    const int wrow = wave >> 1;
    const int wcol = wave & 1;
    const int l15  = lane & 15;
    const int kbase = (lane >> 4) * 8;

    v8f acc[2][4];
    #pragma unroll
    for (int i = 0; i < 2; ++i)
        #pragma unroll
        for (int j = 0; j < 4; ++j) acc[i][j] = {};

    stage_A_async(Abf, m0, 0, As[0], tid);
    stage_B_dw(value, kern, bat, n0, 0, Bs[0], tid);
    async_wait0();
    __syncthreads();

    for (int it = 0; it < NTK; ++it) {
        const int cur = it & 1, nxt = cur ^ 1;
        if (it + 1 < NTK) {
            stage_A_async(Abf, m0, (it + 1) * BK, As[nxt], tid);
            stage_B_dw(value, kern, bat, n0, (it + 1) * BK, Bs[nxt], tid);
        }
        wmma_step(As[cur], Bs[cur], acc, wrow, wcol, l15, kbase);
        async_wait0();
        __syncthreads();
    }

    #pragma unroll
    for (int i = 0; i < 2; ++i) {
        int gmBase = m0 + wrow * 32 + i * 16 + (lane >> 4) * 8;
        #pragma unroll
        for (int j = 0; j < 4; ++j) {
            int gn = n0 + wcol * 64 + j * 16 + l15;
            if (gn < HW) {
                #pragma unroll
                for (int r = 0; r < 8; ++r) {
                    int gm = gmBase + r;
                    Cp[(size_t)gm * HW + gn] = acc[i][j][r] + bias[gm];
                }
            }
        }
    }
}

// ======================================================================
extern "C" void kernel_launch(void* const* d_in, const int* in_sizes, int n_in,
                              void* d_out, int out_size, void* d_ws, size_t ws_size,
                              hipStream_t stream) {
    const float* x      = (const float*)d_in[0];
    const float* w_qk   = (const float*)d_in[1];
    const float* b_qk   = (const float*)d_in[2];
    const float* w_kg1  = (const float*)d_in[3];
    const float* b_kg1  = (const float*)d_in[4];
    const float* w_kg2  = (const float*)d_in[5];
    const float* b_kg2  = (const float*)d_in[6];
    const float* w_v    = (const float*)d_in[7];
    const float* b_v    = (const float*)d_in[8];
    const float* w_proj = (const float*)d_in[9];
    const float* b_proj = (const float*)d_in[10];
    const float* beta   = (const float*)d_in[11];
    float* out = (float*)d_out;

    // workspace layout (all 16B aligned)
    char* ws = (char*)d_ws;
    size_t o = 0;
    float*  xp    = (float*)(ws + o);  o += (size_t)BATCH * CH * NKT * 4;   // 442KB
    float*  kern  = (float*)(ws + o);  o += (size_t)BATCH * CH * NKT * 4;   // 442KB
    bf16_t* wv_bf = (bf16_t*)(ws + o); o += (size_t)CH * CH * 2;            // 288KB
    bf16_t* wp_bf = (bf16_t*)(ws + o); o += (size_t)CH * CH * 2;            // 288KB
    bf16_t* xbf   = (bf16_t*)(ws + o); o += (size_t)BATCH * CH * HW * 2;    // ~72MB
    bf16_t* value = (bf16_t*)(ws + o);                                      // ~72MB

    // 1) pool x -> xp, fused x -> bf16
    pool18_kernel<<<dim3(BATCH * CH), dim3(128), 0, stream>>>(x, xp, xbf);

    // 1b) one-time weight conversion
    cvtw_kernel<<<dim3((CH * CH + 255) / 256), dim3(256), 0, stream>>>(
        w_v, w_proj, wv_bf, wp_bf);

    // 2) kernel-generation MLP -> kern
    kgen_kernel<<<dim3(BATCH), dim3(CH), 0, stream>>>(
        xp, w_qk, b_qk, w_kg1, b_kg1, w_kg2, b_kg2, beta, kern);

    // 3) value GEMM (bf16 WMMA, async A staging, double-buffered)
    const int ntiles = (HW + BN - 1) / BN;   // 23
    gemm_value_kernel<<<dim3(ntiles, CH / BM, BATCH), dim3(256), 0, stream>>>(
        wv_bf, xbf, b_v, value);

    // 4) fused depthwise + proj GEMM (double-buffered)
    gemm_dwproj_kernel<<<dim3(ntiles, CH / BM, BATCH), dim3(256), 0, stream>>>(
        wp_bf, value, kern, b_proj, out);
}
